// TVAAttention_79980880986551
// MI455X (gfx1250) — matrix operand order repaired and checked by reading, not compiled, not run on hardware
//
#include <hip/hip_runtime.h>
#include <hip/hip_bf16.h>

// ---------------------------------------------------------------------------
// TVA (low-rank "vein") attention for MI455X / gfx1250, bf16 WMMA pipeline.
//   B=2, T=2048, D=2048, H=16, dh=128, r=32
// Weight folding:   Weff_{q,k,v}[d, h*32+rr] = sum_i W[h*128+i, d] * Wdown[i, rr]
//                   Woeff[h*32+rr, d]        = sum_i Wup[rr, i]    * Wo[d, h*128+i]
// ---------------------------------------------------------------------------

typedef __attribute__((ext_vector_type(16))) __bf16 v16bf;
typedef __attribute__((ext_vector_type(8)))  __bf16 v8bf;
typedef __attribute__((ext_vector_type(8)))  float  v8f;

#define D_MODEL 2048
#define N_HEADS 16
#define RANK    32
#define CDIM    512          // H * RANK
#define BT      4096         // B * T
#define TSEQ    2048

__device__ inline __bf16 cvt_bf16(float f) {
  unsigned int u = __builtin_bit_cast(unsigned int, f);
  u += 0x7fffu + ((u >> 16) & 1u);      // round-to-nearest-even
  unsigned short h = (unsigned short)(u >> 16);
  return __builtin_bit_cast(__bf16, h);
}

__device__ inline v16bf cat8(v8bf lo, v8bf hi) {
  return __builtin_shufflevector(lo, hi, 0,1,2,3,4,5,6,7,8,9,10,11,12,13,14,15);
}

// A operand (16x32, 16-bit): lane<16 -> row M=lane, K in {0..7, 16..23};
// lane>=16 -> row M=lane-16, K in {8..15, 24..31}.  `rowM` points at [row][kb].
__device__ inline v16bf load_a(const __bf16* rowM, int lane) {
  const int base0 = (lane & 16) ? 8 : 0;
  v8bf lo = *(const v8bf*)(rowM + base0);
  v8bf hi = *(const v8bf*)(rowM + base0 + 16);
  return cat8(lo, hi);
}

// B operand (32x16, 16-bit): lane<16 -> col N=lane, K in {0..7, 16..23};
// lane>=16 -> col N=lane-16, K in {8..15, 24..31}.  `rowN` points at [n][kb] of a [N][K] array.
__device__ inline v16bf load_b(const __bf16* rowN, int lane) {
  const __bf16* p = rowN + ((lane & 16) ? 8 : 0);
  v8bf lo = *(const v8bf*)(p);
  v8bf hi = *(const v8bf*)(p + 16);
  return cat8(lo, hi);
}

__device__ inline v8f wmma_bf16(v16bf a, v16bf b, v8f c) {
  return __builtin_amdgcn_wmma_f32_16x16x32_bf16(false, a, false, b, (short)0, c,
                                                 false, false);
}

// ---------------------------------------------------------------------------
// Weight fusion kernels (tiny, VALU fp32)
// ---------------------------------------------------------------------------
__global__ void fuse_qkv(const float* __restrict__ W, const float* __restrict__ Wdown,
                         __bf16* __restrict__ WT /* [512][2048] */) {
  const int d = blockIdx.x * 256 + threadIdx.x;   // 0..2047
  const int c = blockIdx.y;                       // 0..511
  const int h = c >> 5, rr = c & 31;
  float acc = 0.f;
#pragma unroll 4
  for (int i = 0; i < 128; ++i)
    acc += W[(size_t)(h * 128 + i) * D_MODEL + d] * Wdown[i * RANK + rr];
  WT[(size_t)c * D_MODEL + d] = cvt_bf16(acc);
}

__global__ void fuse_o(const float* __restrict__ Wo, const float* __restrict__ Wup,
                       __bf16* __restrict__ WoT /* [2048][512] */) {
  const int c = blockIdx.x * 256 + threadIdx.x;   // 0..511 (grid.x = 2)
  const int d = blockIdx.y;                       // 0..2047
  const int h = c >> 5, rr = c & 31;
  float acc = 0.f;
#pragma unroll 4
  for (int i = 0; i < 128; ++i)
    acc += Wup[rr * 128 + i] * Wo[(size_t)d * D_MODEL + h * 128 + i];
  WoT[(size_t)d * CDIM + c] = cvt_bf16(acc);
}

// x fp32 -> bf16 (done once, reused by 3 GEMMs)
__global__ void cvt_x(const float* __restrict__ X, __bf16* __restrict__ Xb) {
  const size_t i = ((size_t)blockIdx.x * 256 + threadIdx.x) * 4;
  const float4 v = *(const float4*)(X + i);
  Xb[i + 0] = cvt_bf16(v.x);
  Xb[i + 1] = cvt_bf16(v.y);
  Xb[i + 2] = cvt_bf16(v.z);
  Xb[i + 3] = cvt_bf16(v.w);
}

// ---------------------------------------------------------------------------
// GEMM1: [4096,2048](bf16) x [2048,512](bf16, stored [N][K]) -> rank space
//   out_mode 0: OUT[b][h][t][rr]   (Q_r / K_r)
//   out_mode 1: OUT[b][h][rr][t]   (V_r transposed, for PV B-operand)
// Wave tile: 16 (M) x 64 (N).  Block = 4 waves covering 64 M rows.
// ---------------------------------------------------------------------------
__global__ void gemm_xw(const __bf16* __restrict__ Xb, const __bf16* __restrict__ WT,
                        __bf16* __restrict__ OUT, int out_mode) {
  const int lane = threadIdx.x & 31;
  const int wave = threadIdx.x >> 5;
  const int l16  = lane & 15;
  const int m0   = blockIdx.x * 64 + wave * 16;
  const int n0   = blockIdx.y * 64;

  v8f acc[4] = {};
  const __bf16* arow = Xb + (size_t)(m0 + l16) * D_MODEL;
#pragma unroll 1
  for (int kb = 0; kb < D_MODEL; kb += 32) {
    const v16bf a = load_a(arow + kb, lane);
#pragma unroll
    for (int nc = 0; nc < 4; ++nc) {
      const v16bf b = load_b(WT + (size_t)(n0 + nc * 16 + l16) * D_MODEL + kb, lane);
      acc[nc] = wmma_bf16(a, b, acc[nc]);
    }
  }
  const int rbase = (lane & 16) ? 8 : 0;
#pragma unroll
  for (int nc = 0; nc < 4; ++nc) {
    const int c = n0 + nc * 16 + l16;         // h*32+rr
    const int h = c >> 5, rr = c & 31;
#pragma unroll
    for (int j = 0; j < 8; ++j) {
      const int t  = m0 + rbase + j;          // 0..4095
      const int bb = t >> 11, tt = t & 2047;
      size_t idx;
      if (out_mode == 0)
        idx = ((((size_t)bb * N_HEADS + h) * TSEQ) + tt) * RANK + rr;
      else
        idx = ((((size_t)bb * N_HEADS + h) * RANK) + rr) * TSEQ + tt;
      OUT[idx] = cvt_bf16(acc[nc][j]);
    }
  }
}

// ---------------------------------------------------------------------------
// Flash attention in rank space.  One wave = 16 queries; block = 4 waves.
// grid = (T/64, B*H).  Head dim r=32 == one WMMA K step.
// ---------------------------------------------------------------------------
__global__ void attn_kernel(const __bf16* __restrict__ Qr,   // [BH][T][32]
                            const __bf16* __restrict__ Kr,   // [BH][T][32]
                            const __bf16* __restrict__ VrT,  // [BH][32][T]
                            __bf16* __restrict__ Yr) {       // [B*T][512]
  __shared__ __bf16 plds[4][16][64];                         // P tile per wave, 8 KB
  const int lane  = threadIdx.x & 31;
  const int wave  = threadIdx.x >> 5;
  const int l16   = lane & 15;
  const int half8 = (lane & 16) ? 8 : 0;
  const int bh    = blockIdx.y;                 // b*16 + h
  const int b     = bh >> 4, h = bh & 15;
  const int q0    = blockIdx.x * 64 + wave * 16;
  const float scale = 0.17677669529663687f;     // 1/sqrt(32)

  const __bf16* Qh = Qr  + (size_t)bh * TSEQ * RANK;
  const __bf16* Kh = Kr  + (size_t)bh * TSEQ * RANK;
  const __bf16* Vh = VrT + (size_t)bh * RANK * TSEQ;

  const v16bf qa = load_a(Qh + (size_t)(q0 + l16) * RANK, lane);

  v8f o0 = {}, o1 = {};
  float m_run[8], l_run[8];
#pragma unroll
  for (int j = 0; j < 8; ++j) { m_run[j] = -1e30f; l_run[j] = 0.f; }

  for (int kb = 0; kb < TSEQ; kb += 64) {
    // ---- scores S = Qr * Kr^T  (4 x 16-key chunks) ----
    v8f s[4];
#pragma unroll
    for (int nc = 0; nc < 4; ++nc) {
      const v16bf kB = load_b(Kh + (size_t)(kb + nc * 16 + l16) * RANK, lane);
      v8f z = {};
      s[nc] = wmma_bf16(qa, kB, z);
    }
    // ---- online softmax (row j lives in 16-lane half; xor 1/2/4/8 reduces it) ----
    float mnew[8], alpha[8];
#pragma unroll
    for (int j = 0; j < 8; ++j) {
      float mx = fmaxf(fmaxf(s[0][j], s[1][j]), fmaxf(s[2][j], s[3][j])) * scale;
      for (int d = 1; d < 16; d <<= 1) mx = fmaxf(mx, __shfl_xor(mx, d, 32));
      mnew[j]  = fmaxf(m_run[j], mx);
      alpha[j] = __expf(m_run[j] - mnew[j]);
      m_run[j] = mnew[j];
    }
#pragma unroll
    for (int j = 0; j < 8; ++j) {
      float rs = 0.f;
#pragma unroll
      for (int nc = 0; nc < 4; ++nc) {
        const float p = __expf(s[nc][j] * scale - mnew[j]);
        plds[wave][half8 + j][nc * 16 + l16] = cvt_bf16(p);
        rs += p;
      }
      for (int d = 1; d < 16; d <<= 1) rs += __shfl_xor(rs, d, 32);
      l_run[j] = l_run[j] * alpha[j] + rs;
      o0[j] = o0[j] * alpha[j];
      o1[j] = o1[j] * alpha[j];
    }
    // order ds_store (C layout) before ds_load (A layout); same-wave LDS is in-order
    __asm__ volatile("s_wait_dscnt 0x0" ::: "memory");

    // ---- y_r += P * V_r  (A = P from LDS, B = VrT rows) ----
    const __bf16* prow = &plds[wave][l16][0];
#pragma unroll
    for (int ks = 0; ks < 2; ++ks) {
      const v16bf pa  = load_a(prow + ks * 32, lane);
      const v16bf vb0 = load_b(Vh + (size_t)(l16)      * TSEQ + kb + ks * 32, lane);
      const v16bf vb1 = load_b(Vh + (size_t)(16 + l16) * TSEQ + kb + ks * 32, lane);
      o0 = wmma_bf16(pa, vb0, o0);
      o1 = wmma_bf16(pa, vb1, o1);
    }
  }
  // ---- normalize and emit y_r as [b*T+t][h*32+rr] bf16 ----
#pragma unroll
  for (int j = 0; j < 8; ++j) {
    const float inv = 1.0f / l_run[j];
    const int t = q0 + half8 + j;
    const size_t row = ((size_t)b * TSEQ + t) * CDIM + h * RANK;
    Yr[row + l16]      = cvt_bf16(o0[j] * inv);
    Yr[row + 16 + l16] = cvt_bf16(o1[j] * inv);
  }
}

// ---------------------------------------------------------------------------
// GEMM2: [4096,512](bf16) x [512,2048](bf16, stored [N][K]=[2048][512]) -> fp32 out
// ---------------------------------------------------------------------------
__global__ void gemm_yo(const __bf16* __restrict__ Y, const __bf16* __restrict__ WoT,
                        float* __restrict__ OUT) {
  const int lane = threadIdx.x & 31;
  const int wave = threadIdx.x >> 5;
  const int l16  = lane & 15;
  const int m0   = blockIdx.x * 64 + wave * 16;
  const int n0   = blockIdx.y * 64;

  v8f acc[4] = {};
  const __bf16* arow = Y + (size_t)(m0 + l16) * CDIM;
#pragma unroll 1
  for (int kb = 0; kb < CDIM; kb += 32) {
    const v16bf a = load_a(arow + kb, lane);
#pragma unroll
    for (int nc = 0; nc < 4; ++nc) {
      const v16bf b = load_b(WoT + (size_t)(n0 + nc * 16 + l16) * CDIM + kb, lane);
      acc[nc] = wmma_bf16(a, b, acc[nc]);
    }
  }
  const int rbase = (lane & 16) ? 8 : 0;
#pragma unroll
  for (int nc = 0; nc < 4; ++nc) {
#pragma unroll
    for (int j = 0; j < 8; ++j)
      OUT[(size_t)(m0 + rbase + j) * D_MODEL + n0 + nc * 16 + l16] = acc[nc][j];
  }
}

// ---------------------------------------------------------------------------
extern "C" void kernel_launch(void* const* d_in, const int* in_sizes, int n_in,
                              void* d_out, int out_size, void* d_ws, size_t ws_size,
                              hipStream_t stream) {
  (void)in_sizes; (void)n_in; (void)out_size; (void)ws_size;
  const float* x     = (const float*)d_in[0];
  const float* Wq    = (const float*)d_in[1];
  const float* Wk    = (const float*)d_in[2];
  const float* Wv    = (const float*)d_in[3];
  const float* Wo    = (const float*)d_in[4];
  const float* Wdown = (const float*)d_in[5];
  const float* Wup   = (const float*)d_in[6];
  float* out = (float*)d_out;

  // workspace layout (bf16 elements)
  __bf16* WqT = (__bf16*)d_ws;                       // [512][2048]
  __bf16* WkT = WqT + (size_t)CDIM * D_MODEL;
  __bf16* WvT = WkT + (size_t)CDIM * D_MODEL;
  __bf16* WoT = WvT + (size_t)CDIM * D_MODEL;        // [2048][512]
  __bf16* Xb  = WoT + (size_t)D_MODEL * CDIM;        // [4096][2048]
  __bf16* Qr  = Xb  + (size_t)BT * D_MODEL;          // [32][2048][32]
  __bf16* Kr  = Qr  + (size_t)32 * TSEQ * RANK;
  __bf16* VrT = Kr  + (size_t)32 * TSEQ * RANK;      // [32][32][2048]
  __bf16* Yr  = VrT + (size_t)32 * RANK * TSEQ;      // [4096][512]

  const dim3 b256(256), b128(128);
  fuse_qkv<<<dim3(8, 512),  b256, 0, stream>>>(Wq, Wdown, WqT);
  fuse_qkv<<<dim3(8, 512),  b256, 0, stream>>>(Wk, Wdown, WkT);
  fuse_qkv<<<dim3(8, 512),  b256, 0, stream>>>(Wv, Wdown, WvT);
  fuse_o  <<<dim3(2, 2048), b256, 0, stream>>>(Wo, Wup, WoT);
  cvt_x   <<<dim3((BT * D_MODEL) / (256 * 4)), b256, 0, stream>>>(x, Xb);

  gemm_xw<<<dim3(64, 8), b128, 0, stream>>>(Xb, WqT, Qr, 0);
  gemm_xw<<<dim3(64, 8), b128, 0, stream>>>(Xb, WkT, Kr, 0);
  gemm_xw<<<dim3(64, 8), b128, 0, stream>>>(Xb, WvT, VrT, 1);

  attn_kernel<<<dim3(32, 32), b128, 0, stream>>>(Qr, Kr, VrT, Yr);

  gemm_yo<<<dim3(64, 32), b128, 0, stream>>>(Yr, WoT, out);
}
